// my_bidirectional_lstm_14542759264271
// MI455X (gfx1250) — compile-verified, matrix operand order checked by reading, not run on hardware
//
#include <hip/hip_runtime.h>
#include <hip/hip_bf16.h>

// ---- problem constants (from reference) ----
static constexpr int T_   = 256;   // SEQ
static constexpr int NB   = 64;    // batch
static constexpr int NI   = 256;   // input size
static constexpr int NH   = 1024;  // hidden
static constexpr int NG   = 4096;  // 4*H gates
static constexpr int KC   = NI + NH;   // 1280 combined K for fused step GEMM
static constexpr int NOUT = 1024;
static constexpr int KFC  = 2 * NH;    // 2048

typedef __attribute__((ext_vector_type(16))) __bf16 v16bf;
typedef __attribute__((ext_vector_type(8)))  __bf16 v8bf;
typedef __attribute__((ext_vector_type(8)))  float  v8f;

// Load one 16-bit WMMA operand fragment for this lane.
// Per CDNA5 ISA 7.12.2 (16-bit A 16x32 / B 32x16): a lane holds 8 contiguous
// elements at its K-base and 8 more at K-base+16. p points at element K-base
// of the lane's row (A: row m = lane&15; B from pre-transposed W: row n = lane&15).
__device__ inline v16bf ld_frag(const __bf16* __restrict__ p) {
  v8bf lo = *reinterpret_cast<const v8bf*>(p);
  v8bf hi = *reinterpret_cast<const v8bf*>(p + 16);
  v16bf r;
#pragma unroll
  for (int e = 0; e < 8; ++e) { r[e] = lo[e]; r[e + 8] = hi[e]; }
  return r;
}

// fast sigmoid / tanh on v_exp_f32; saturate correctly at +-inf
__device__ inline float sigm(float x) { return 1.0f / (1.0f + __expf(-x)); }
__device__ inline float tanh_fast(float x) { return 2.0f * sigm(2.0f * x) - 1.0f; }

// ---- one-time conversion / layout kernels ----

__global__ void cvt_x_kernel(const float* __restrict__ x, __bf16* __restrict__ xb) {
  int i = blockIdx.x * blockDim.x + threadIdx.x;       // exact grid, no bounds
  xb[i] = (__bf16)x[i];
}

// Wcat^T[n][k] = k < NI ? Wx[k][n] : Wh[k-NI][n]   -> [NG, KC] bf16 row-major
__global__ void build_wcat_kernel(const float* __restrict__ Wx,
                                  const float* __restrict__ Wh,
                                  __bf16* __restrict__ Wt) {
  int i = blockIdx.x * blockDim.x + threadIdx.x;       // NG*KC threads exactly
  int n = i / KC;
  int k = i - n * KC;
  float v = (k < NI) ? Wx[(size_t)k * NG + n] : Wh[(size_t)(k - NI) * NG + n];
  Wt[i] = (__bf16)v;
}

// Wfc^T[o][k] = W_fc[k][o]  -> [NOUT, KFC] bf16 row-major
__global__ void build_wfc_kernel(const float* __restrict__ Wfc, __bf16* __restrict__ Wt) {
  int i = blockIdx.x * blockDim.x + threadIdx.x;       // NOUT*KFC threads
  int n = i / KFC;
  int k = i - n * KFC;
  Wt[i] = (__bf16)Wfc[(size_t)k * NOUT + n];
}

// ---- fused recurrent step: g = [x_t|h_prev] @ Wcat + b ; cell update ----
// One launch per timestep s; handles forward t=s and backward t=T-1-s.
// 512 wave-jobs: dir(2) x m_tiles(4) x h_tiles(64). Each wave: 4 gate
// accumulators sharing the A fragment -> 4 v_wmma per K-chunk. K loop is
// split x-part / h-part so all addresses advance linearly (no cndmask).
__global__ void __launch_bounds__(128)
lstm_step_kernel(const __bf16* __restrict__ xb,
                 const __bf16* __restrict__ WcatTf, const __bf16* __restrict__ WcatTb,
                 const float* __restrict__ b_f, const float* __restrict__ b_b,
                 __bf16* __restrict__ hs_f, __bf16* __restrict__ hs_b,
                 float* __restrict__ c_f, float* __restrict__ c_b,
                 float* __restrict__ hl_f, float* __restrict__ hl_b,
                 int s) {
  const int lane = threadIdx.x & 31;
  const int wave = blockIdx.x * (blockDim.x >> 5) + (threadIdx.x >> 5);
  const int dir  = wave >> 8;          // 0 = fwd, 1 = bwd
  const int id   = wave & 255;
  const int m_tile = (id >> 6) << 4;   // 0,16,32,48 (batch)
  const int h_tile = (id & 63) << 4;   // 0..1008   (hidden)

  const int  t     = dir ? (T_ - 1 - s) : s;
  const bool first = (s == 0);

  const __bf16* Wt = dir ? WcatTb : WcatTf;
  const float*  bb = dir ? b_b : b_f;
  __bf16* hs = dir ? hs_b : hs_f;
  float*  c  = dir ? c_b : c_f;
  float*  hl = dir ? hl_b : hl_f;

  const int lanek = (lane >> 4) << 3;  // 0 or 8: this lane's K sub-base
  const int mrow  = m_tile + (lane & 15);
  const int ncol  = lane & 15;

  // lane-resident row base pointers
  const __bf16* xrow = xb + (size_t)t * (NB * NI) + (size_t)mrow * NI + lanek;
  const __bf16* wrow = Wt + (size_t)(h_tile + ncol) * KC + lanek;  // gate 0 row

  v8f acc[4];
#pragma unroll
  for (int g = 0; g < 4; ++g)
#pragma unroll
    for (int e = 0; e < 8; ++e) acc[g][e] = 0.0f;

  // ---- x-part: K = 0 .. NI ----
#pragma unroll 2
  for (int ci = 0; ci < NI / 32; ++ci) {
    const int k0 = ci * 32;
    v16bf a = ld_frag(xrow + k0);
#pragma unroll
    for (int g = 0; g < 4; ++g) {
      v16bf bfr = ld_frag(wrow + (size_t)g * (NH * KC) + k0);
      acc[g] = __builtin_amdgcn_wmma_f32_16x16x32_bf16(
          false, a, false, bfr, (short)0, acc[g], false, false);
    }
  }

  // ---- h-part: K = NI .. KC (h_prev == 0 at s == 0) ----
  if (!first) {
    const __bf16* hrow = hs + (size_t)(dir ? t + 1 : t - 1) * (NB * NH)
                            + (size_t)mrow * NH + lanek;
    const __bf16* wrowh = wrow + NI;
#pragma unroll 2
    for (int ci = 0; ci < NH / 32; ++ci) {
      const int k0 = ci * 32;
      v16bf a = ld_frag(hrow + k0);
#pragma unroll
      for (int g = 0; g < 4; ++g) {
        v16bf bfr = ld_frag(wrowh + (size_t)g * (NH * KC) + k0);
        acc[g] = __builtin_amdgcn_wmma_f32_16x16x32_bf16(
            false, a, false, bfr, (short)0, acc[g], false, false);
      }
    }
  }

  // --- fused cell update (f32 math) ---
  const int   hcol = h_tile + ncol;
  const float bi = bb[0 * NH + hcol];
  const float bf = bb[1 * NH + hcol];
  const float bo = bb[2 * NH + hcol];
  const float bc = bb[3 * NH + hcol];
  __bf16* hs_t  = hs + (size_t)t * (NB * NH);
  const int mbase = m_tile + ((lane >> 4) << 3);   // C/D layout: m = r + 8*(lane/16)
#pragma unroll
  for (int r = 0; r < 8; ++r) {
    const int    m   = mbase + r;
    const size_t idx = (size_t)m * NH + hcol;
    float gi = sigm(acc[0][r] + bi);
    float gf = sigm(acc[1][r] + bf);
    float go = sigm(acc[2][r] + bo);
    float gc = tanh_fast(acc[3][r] + bc);
    float cold = first ? 0.0f : c[idx];
    float cn = gf * cold + gi * gc;
    float hn = go * tanh_fast(cn);
    c[idx]    = cn;
    hs_t[idx] = (__bf16)hn;   // bf16 for next step's GEMM and the FC
    hl[idx]   = hn;           // f32 final-state capture (last write wins)
  }
}

// ---- final FC: [T*B, 2H] @ Wfc + b_fc -> d_out ----
// A row r = t*NB+b; k<NH from hs_f, else hs_b. 16384 wave-jobs, each wave
// computes 4 adjacent N tiles reusing the A fragment. K loop split per source.
__global__ void __launch_bounds__(256)
fc_kernel(const __bf16* __restrict__ hs_f, const __bf16* __restrict__ hs_b,
          const __bf16* __restrict__ WfcT, const float* __restrict__ b_fc,
          float* __restrict__ out) {
  const int lane = threadIdx.x & 31;
  const int wave = blockIdx.x * (blockDim.x >> 5) + (threadIdx.x >> 5);
  const int m_tile = (wave >> 4) << 4;     // 1024 row tiles
  const int n_base = (wave & 15) << 6;     // 16 groups of 64 cols

  const int lanek = (lane >> 4) << 3;
  const int mrow  = m_tile + (lane & 15);
  const int ncol  = lane & 15;

  const __bf16* frow = hs_f + (size_t)mrow * NH + lanek;
  const __bf16* brow = hs_b + (size_t)mrow * NH + lanek;
  const __bf16* wrow = WfcT + (size_t)(n_base + ncol) * KFC + lanek;

  v8f acc[4];
#pragma unroll
  for (int j = 0; j < 4; ++j)
#pragma unroll
    for (int e = 0; e < 8; ++e) acc[j][e] = 0.0f;

#pragma unroll 2
  for (int ci = 0; ci < NH / 32; ++ci) {
    const int k0 = ci * 32;
    v16bf a = ld_frag(frow + k0);
#pragma unroll
    for (int j = 0; j < 4; ++j) {
      v16bf bfr = ld_frag(wrow + (size_t)j * (16 * KFC) + k0);
      acc[j] = __builtin_amdgcn_wmma_f32_16x16x32_bf16(
          false, a, false, bfr, (short)0, acc[j], false, false);
    }
  }
#pragma unroll 2
  for (int ci = 0; ci < NH / 32; ++ci) {
    const int k0 = ci * 32;
    v16bf a = ld_frag(brow + k0);
#pragma unroll
    for (int j = 0; j < 4; ++j) {
      v16bf bfr = ld_frag(wrow + (size_t)j * (16 * KFC) + NH + k0);
      acc[j] = __builtin_amdgcn_wmma_f32_16x16x32_bf16(
          false, a, false, bfr, (short)0, acc[j], false, false);
    }
  }

  const int mbase = m_tile + ((lane >> 4) << 3);
#pragma unroll
  for (int j = 0; j < 4; ++j) {
    const int n  = n_base + j * 16 + ncol;
    const float bn = b_fc[n];
#pragma unroll
    for (int r = 0; r < 8; ++r)
      out[(size_t)(mbase + r) * NOUT + n] = acc[j][r] + bn;
  }
}

// ---- final states: (outputs, hF, hB, cF, cB) flat in return order ----
__global__ void tail_kernel(const float* __restrict__ hlf, const float* __restrict__ hlb,
                            const float* __restrict__ cf,  const float* __restrict__ cb,
                            float* __restrict__ out) {
  int i = blockIdx.x * blockDim.x + threadIdx.x;       // NB*NH threads exactly
  const size_t base = (size_t)T_ * NB * NOUT;
  const size_t st   = (size_t)NB * NH;
  out[base + 0 * st + i] = hlf[i];
  out[base + 1 * st + i] = hlb[i];
  out[base + 2 * st + i] = cf[i];
  out[base + 3 * st + i] = cb[i];
}

extern "C" void kernel_launch(void* const* d_in, const int* in_sizes, int n_in,
                              void* d_out, int out_size, void* d_ws, size_t ws_size,
                              hipStream_t stream) {
  const float* x    = (const float*)d_in[0];
  const float* Wx_f = (const float*)d_in[1];
  const float* Wh_f = (const float*)d_in[2];
  const float* b_f  = (const float*)d_in[3];
  const float* Wx_b = (const float*)d_in[4];
  const float* Wh_b = (const float*)d_in[5];
  const float* b_b  = (const float*)d_in[6];
  const float* W_fc = (const float*)d_in[7];
  const float* b_fc = (const float*)d_in[8];
  float* out = (float*)d_out;

  // ---- workspace layout (bytes) ----
  char* ws = (char*)d_ws;
  __bf16* xb     = (__bf16*)(ws);                        //  8,388,608  x in bf16
  __bf16* WcatTf = (__bf16*)(ws +   8388608u);           // 10,485,760  [NG,KC]
  __bf16* WcatTb = (__bf16*)(ws +  18874368u);           // 10,485,760
  __bf16* WfcT   = (__bf16*)(ws +  29360128u);           //  4,194,304  [NOUT,KFC]
  __bf16* hs_f   = (__bf16*)(ws +  33554432u);           // 33,554,432  [T,NB,NH]
  __bf16* hs_b   = (__bf16*)(ws +  67108864u);           // 33,554,432
  float*  c_f    = (float*) (ws + 100663296u);           //    262,144
  float*  c_b    = (float*) (ws + 100925440u);           //    262,144
  float*  hl_f   = (float*) (ws + 101187584u);           //    262,144
  float*  hl_b   = (float*) (ws + 101449728u);           //    262,144  -> ~101.7 MB

  // one-time precision conversion + weight transpose (weights then live in L2)
  cvt_x_kernel<<<(T_ * NB * NI) / 256, 256, 0, stream>>>(x, xb);
  build_wcat_kernel<<<(NG * KC) / 256, 256, 0, stream>>>(Wx_f, Wh_f, WcatTf);
  build_wcat_kernel<<<(NG * KC) / 256, 256, 0, stream>>>(Wx_b, Wh_b, WcatTb);
  build_wfc_kernel<<<(NOUT * KFC) / 256, 256, 0, stream>>>(W_fc, WfcT);

  // sequential scan: one fused GEMM+cell kernel per timestep (both directions)
  for (int s = 0; s < T_; ++s)
    lstm_step_kernel<<<128, 128, 0, stream>>>(xb, WcatTf, WcatTb, b_f, b_b,
                                              hs_f, hs_b, c_f, c_b, hl_f, hl_b, s);

  // output projection + bias, then final states
  fc_kernel<<<2048, 256, 0, stream>>>(hs_f, hs_b, WfcT, b_fc, out);
  tail_kernel<<<(NB * NH) / 256, 256, 0, stream>>>(hl_f, hl_b, c_f, c_b, out);
}
